// MemoryEfficientAttention_19954418057475
// MI455X (gfx1250) — matrix operand order repaired and checked by reading, not compiled, not run on hardware
//
#include <hip/hip_runtime.h>
#include <hip/hip_bf16.h>
#include <math.h>

// ---------------------------------------------------------------------------
// Problem constants (from reference): B=2, N=2048, C=1024, H=16, D=64
// ---------------------------------------------------------------------------
#define BB    2
#define NN    2048
#define CC    1024
#define HH    16
#define DD    64
#define C3    (3*CC)          // 3072
#define MROWS (BB*NN)         // 4096
#define ATTN_SCALE 0.125f     // 64^-0.5

typedef __bf16 bf16_t;
typedef bf16_t v8bf  __attribute__((ext_vector_type(8)));
typedef bf16_t v16bf __attribute__((ext_vector_type(16)));
typedef float  v8f   __attribute__((ext_vector_type(8)));

#define DEVINL static __device__ __forceinline__

DEVINL v16bf cat8(v8bf a, v8bf b) {
  return __builtin_shufflevector(a, b, 0,1,2,3,4,5,6,7,8,9,10,11,12,13,14,15);
}

DEVINL v8f wmma_bf16(v16bf a, v16bf b, v8f c) {
  // D(f32 16x16) = A(bf16 16x32) * B(bf16 32x16) + C
  return __builtin_amdgcn_wmma_f32_16x16x32_bf16(
      /*neg_a=*/false, a, /*neg_b=*/false, b,
      /*c_mod=*/(short)0, c, /*reuse_a=*/false, /*reuse_b=*/false);
}

// A fragment 16(M) x 32(K) from row-major storage with row stride (elements).
// ISA layout: lane L (<16): row M=L, K {0..7,16..23}; lane 16+m: row m, K {8..15,24..31}
DEVINL v16bf load_a_frag(const bf16_t* base, int stride) {
  int lane = threadIdx.x & 31;
  int m    = lane & 15;
  int koff = (lane >> 4) << 3;
  const bf16_t* p = base + (size_t)m * stride + koff;
  v8bf lo = *(const v8bf*)(p);
  v8bf hi = *(const v8bf*)(p + 16);
  return cat8(lo, hi);
}

// B fragment 32(K) x 16(N): lane L holds column N=L%16, K {0..7,16..23} (L<16) / {8..15,24..31}
DEVINL v16bf load_b_frag(const bf16_t* base, int stride) {
  int lane = threadIdx.x & 31;
  int n    = lane & 15;
  int kh   = (lane >> 4) << 3;
  v16bf t;
#pragma unroll
  for (int s = 0; s < 16; s++) {
    int k = (s < 8) ? (kh + s) : (16 + kh + (s - 8));
    t[s] = base[(size_t)k * stride + n];
  }
  return t;
}

// ---- CDNA5 async global->LDS copy (ASYNCcnt path), 16B per lane -----------
// vdst = wave-relative LDS byte offset, vaddr = 64-bit global address.
DEVINL void async_ld_b128(const bf16_t* g, bf16_t* l) {
  unsigned lds_off = (unsigned)(size_t)l;   // low 32 bits of LDS aperture addr
  asm volatile("global_load_async_to_lds_b128 %0, %1, off"
               :: "v"(lds_off), "v"(g) : "memory");
}
DEVINL void wait_async0() {
  asm volatile("s_wait_asynccnt 0x0" ::: "memory");
}

// ---------------------------------------------------------------------------
// fp32 -> bf16 cast (n divisible by 4)
// ---------------------------------------------------------------------------
__global__ __launch_bounds__(256)
void cast_f32_bf16_kernel(const float* __restrict__ in, bf16_t* __restrict__ out, int n) {
  int i = (blockIdx.x * 256 + threadIdx.x) * 4;
  if (i < n) {
    float4 v = *(const float4*)(in + i);
    out[i + 0] = (bf16_t)v.x;
    out[i + 1] = (bf16_t)v.y;
    out[i + 2] = (bf16_t)v.z;
    out[i + 3] = (bf16_t)v.w;
  }
}

// ---------------------------------------------------------------------------
// bf16 GEMM: C[M,Nc] = A[M,K] * B[K,Nc]  (128x128 tile, 8 waves, K-step 32)
// Double-buffered LDS tiles filled by async global->LDS loads: tile k+1
// streams in (ASYNCcnt) while WMMAs consume tile k; one barrier/iteration.
// FINAL=false: store bf16.  FINAL=true: store f32 + bias.
// ---------------------------------------------------------------------------
template <bool FINAL>
__global__ __launch_bounds__(256)
void gemm_bf16_kernel(const bf16_t* __restrict__ A, const bf16_t* __restrict__ Bm,
                      int M, int Nc, int K,
                      bf16_t* __restrict__ Cb, float* __restrict__ Cf,
                      const float* __restrict__ bias) {
  __shared__ bf16_t As[2][128 * 32];
  __shared__ bf16_t Bs[2][32 * 128];

  const int tid  = threadIdx.x;
  const int wave = tid >> 5;
  const int lane = tid & 31;
  const int m0   = blockIdx.y * 128;
  const int n0   = blockIdx.x * 128;

  // Per-thread staging coordinates: 16 halfs (2 x b128) for A and for B
  const int idx = tid * 16;
  const int ra = idx >> 5,  ca = idx & 31;    // A tile [128][32]
  const int rb = idx >> 7,  cb = idx & 127;   // B tile [32][128]

  v8f acc[8] = {};

  // Prologue: async-fill buffer 0 with K-step 0
  {
    const bf16_t* ga = A  + (size_t)(m0 + ra) * K  + ca;
    const bf16_t* gb = Bm + (size_t)(rb) * Nc + n0 + cb;
    async_ld_b128(ga,     &As[0][idx]);
    async_ld_b128(ga + 8, &As[0][idx + 8]);
    async_ld_b128(gb,     &Bs[0][idx]);
    async_ld_b128(gb + 8, &Bs[0][idx + 8]);
  }
  wait_async0();
  __syncthreads();

  const int KSTEPS = K >> 5;
  for (int ks = 0; ks < KSTEPS; ks++) {
    const int cur = ks & 1;
    const bool more = (ks + 1) < KSTEPS;

    // Stream next tile into the other buffer while computing this one
    if (more) {
      const int nxt = cur ^ 1;
      const int k1 = (ks + 1) << 5;
      const bf16_t* ga = A  + (size_t)(m0 + ra) * K  + k1 + ca;
      const bf16_t* gb = Bm + (size_t)(k1 + rb) * Nc + n0 + cb;
      async_ld_b128(ga,     &As[nxt][idx]);
      async_ld_b128(ga + 8, &As[nxt][idx + 8]);
      async_ld_b128(gb,     &Bs[nxt][idx]);
      async_ld_b128(gb + 8, &Bs[nxt][idx + 8]);
    }

    v16bf af = load_a_frag(&As[cur][wave * 16 * 32], 32);
#pragma unroll
    for (int j = 0; j < 8; j++) {
      v16bf bf = load_b_frag(&Bs[cur][j * 16], 128);
      acc[j] = wmma_bf16(af, bf, acc[j]);
    }

    if (more) wait_async0();
    __syncthreads();
  }

  // Store: acc[j] element r -> row = wave*16 + r + 8*(lane>=16), col = n0 + j*16 + (lane&15)
  const int mbase = m0 + wave * 16 + ((lane >> 4) << 3);
  const int nbase = n0 + (lane & 15);
#pragma unroll
  for (int j = 0; j < 8; j++) {
#pragma unroll
    for (int r = 0; r < 8; r++) {
      size_t off = (size_t)(mbase + r) * Nc + nbase + j * 16;
      if (FINAL) Cf[off] = acc[j][r] + bias[nbase + j * 16];
      else       Cb[off] = (bf16_t)acc[j][r];
    }
  }
}

// ---------------------------------------------------------------------------
// Fused flash attention. One wave = one 16-query tile of one (b,h).
// qkv: [4096 rows][3072] bf16 (row = b*N + n; cols: q | k | v, each h*64+d)
// out: [4096][1024] bf16 (== [B,N,C] merged-head layout)
// Computes S^T = K*Q^T per 32-key chunk (contiguous A loads for K), per-lane
// online softmax over keys, P staged via LDS into A-fragment form, O += P*V.
// ---------------------------------------------------------------------------
__global__ __launch_bounds__(128)
void attn_kernel(const bf16_t* __restrict__ qkv, bf16_t* __restrict__ out) {
  __shared__ bf16_t lds[4 * (16 * 64 + 16 * 32)];  // per-wave: Q tile + P tile

  const int wave = threadIdx.x >> 5;
  const int lane = threadIdx.x & 31;
  bf16_t* Qs = lds + wave * (16 * 64 + 16 * 32);
  bf16_t* Ps = Qs + 16 * 64;

  const int wid = blockIdx.x * 4 + wave;
  const int qt  = wid & 127;          // query tile within head (N/16 = 128)
  const int h   = (wid >> 7) & 15;
  const int b   = wid >> 11;
  const int q0  = qt * 16;

  const bf16_t* qbase = qkv + (size_t)b * NN * C3 + h * DD;   // + n*C3 + d
  const bf16_t* kbase = qbase + CC;
  const bf16_t* vbase = qbase + 2 * CC;

  // ---- Stage Q tile [16][64] to LDS: each lane 32 contiguous halfs (64B) ----
  {
    int idx = lane * 32;
    int r = idx >> 6, c = idx & 63;
    const bf16_t* src = qbase + (size_t)(q0 + r) * C3 + c;
#pragma unroll
    for (int i = 0; i < 4; i++)
      *(v8bf*)(&Qs[idx + 8 * i]) = *(const v8bf*)(src + 8 * i);
  }
  __syncthreads();

  // ---- Build Q^T B-fragments [32 d x 16 q] once (lane = q column, d in table order) ----
  v16bf qtf[2];
#pragma unroll
  for (int dd = 0; dd < 2; dd++) {
    int q = lane & 15;
    int kh = (lane >> 4) << 3;
    v16bf t;
#pragma unroll
    for (int s = 0; s < 16; s++) t[s] = Qs[q * 64 + dd * 32 + ((s < 8) ? (kh + s) : (16 + kh + s - 8))];
    qtf[dd] = t;
  }

  v8f   o[4] = {};                 // O tile 16q x 64d (4 N-frags)
  float m_run = -1.0e30f, l_run = 0.0f;

  for (int n0k = 0; n0k < NN; n0k += 32) {
    // prefetch next chunk's K rows (global_prefetch_b8)
    if (n0k + 32 < NN)
      __builtin_prefetch(kbase + (size_t)(n0k + 32 + lane) * C3, 0, 1);

    // ---- S^T[32 keys x 16 q] = K * Q^T ----
    v8f s[2] = {};
#pragma unroll
    for (int kk = 0; kk < 2; kk++)
#pragma unroll
      for (int dd = 0; dd < 2; dd++) {
        v16bf kf = load_a_frag(kbase + (size_t)(n0k + kk * 16) * C3 + dd * 32, C3);
        s[kk] = wmma_bf16(kf, qtf[dd], s[kk]);
      }

    // ---- online softmax (per query == per lane column, halves via xor16) ----
    float cmax = -1.0e30f;
#pragma unroll
    for (int kk = 0; kk < 2; kk++)
#pragma unroll
      for (int r = 0; r < 8; r++) {
        float sv = s[kk][r] * ATTN_SCALE;
        s[kk][r] = sv;
        cmax = fmaxf(cmax, sv);
      }
    cmax = fmaxf(cmax, __shfl_xor(cmax, 16, 32));
    float mnew = fmaxf(m_run, cmax);
    float fold = __expf(m_run - mnew);
    float csum = 0.0f;
#pragma unroll
    for (int kk = 0; kk < 2; kk++)
#pragma unroll
      for (int r = 0; r < 8; r++) {
        float p = __expf(s[kk][r] - mnew);
        s[kk][r] = p;
        csum += p;
      }
    csum += __shfl_xor(csum, 16, 32);
    l_run = l_run * fold + csum;
    m_run = mnew;

    // rescale O rows: row r holds query r (lanes<16) / 8+r (lanes>=16)
#pragma unroll
    for (int r = 0; r < 8; r++) {
      float fr = __shfl(fold, (lane < 16) ? r : (8 + r), 32);
#pragma unroll
      for (int j = 0; j < 4; j++) o[j][r] *= fr;
    }

    // ---- stage P (bf16) to LDS as [16 q][32 keys] ----
    {
      int q = lane & 15;
      int kadd = (lane >> 4) << 3;
#pragma unroll
      for (int kk = 0; kk < 2; kk++)
#pragma unroll
        for (int r = 0; r < 8; r++)
          Ps[q * 32 + kk * 16 + r + kadd] = (bf16_t)s[kk][r];
    }
    __syncthreads();

    // ---- O += P * V ----
    v16bf pf = load_a_frag(Ps, 32);
#pragma unroll
    for (int j = 0; j < 4; j++) {
      v16bf vf = load_b_frag(vbase + (size_t)n0k * C3 + j * 16, C3);
      o[j] = wmma_bf16(pf, vf, o[j]);
    }
    __syncthreads();
  }

  // ---- normalize and store O (bf16) into [4096][1024] ----
  float linv = 1.0f / l_run;
#pragma unroll
  for (int r = 0; r < 8; r++) {
    float fr = __shfl(linv, (lane < 16) ? r : (8 + r), 32);
    int qq = q0 + r + ((lane >> 4) << 3);
    size_t rowoff = ((size_t)b * NN + qq) * CC + h * DD + (lane & 15);
#pragma unroll
    for (int j = 0; j < 4; j++)
      out[rowoff + j * 16] = (bf16_t)(o[j][r] * fr);
  }
}

// ---------------------------------------------------------------------------
// Host launcher
// ---------------------------------------------------------------------------
extern "C" void kernel_launch(void* const* d_in, const int* in_sizes, int n_in,
                              void* d_out, int out_size, void* d_ws, size_t ws_size,
                              hipStream_t stream) {
  (void)in_sizes; (void)n_in; (void)out_size; (void)ws_size;

  const float* x      = (const float*)d_in[0];   // [2,2048,1024]
  const float* w_qkv  = (const float*)d_in[1];   // [1024,3072]
  const float* w_proj = (const float*)d_in[2];   // [1024,1024]
  const float* b_proj = (const float*)d_in[3];   // [1024]
  float* outp         = (float*)d_out;           // [2,2048,1024]

  // Workspace layout (bf16 buffers), 256B-aligned offsets
  char* w = (char*)d_ws;
  bf16_t* xb     = (bf16_t*)(w);                          //  8 MB: x bf16
  bf16_t* wqkvb  = (bf16_t*)(w + 8388608);                //  6 MB
  bf16_t* wprojb = (bf16_t*)(w + 14680064);               //  2 MB
  bf16_t* qkvb   = (bf16_t*)(w + 16777216);               // 24 MB: [4096][3072]
  bf16_t* attnb  = (bf16_t*)(w + 41943040);               //  8 MB: [4096][1024]

  const int nx = MROWS * CC;       // 4,194,304
  const int nq = CC * C3;          // 3,145,728
  const int np = CC * CC;          // 1,048,576

  cast_f32_bf16_kernel<<<(nx / 4 + 255) / 256, 256, 0, stream>>>(x, xb, nx);
  cast_f32_bf16_kernel<<<(nq / 4 + 255) / 256, 256, 0, stream>>>(w_qkv, wqkvb, nq);
  cast_f32_bf16_kernel<<<(np / 4 + 255) / 256, 256, 0, stream>>>(w_proj, wprojb, np);

  // qkv = x @ w_qkv : M=4096, N=3072, K=1024 -> bf16
  gemm_bf16_kernel<false><<<dim3(C3 / 128, MROWS / 128), 256, 0, stream>>>(
      xb, wqkvb, MROWS, C3, CC, qkvb, nullptr, nullptr);

  // fused attention: B*H*(N/16) = 4096 waves, 4 waves/block
  attn_kernel<<<(BB * HH * (NN / 16)) / 4, 128, 0, stream>>>(qkvb, attnb);

  // out = attn @ w_proj + b_proj : M=4096, N=1024, K=1024 -> f32
  gemm_bf16_kernel<true><<<dim3(CC / 128, MROWS / 128), 256, 0, stream>>>(
      attnb, wprojb, MROWS, CC, CC, nullptr, outp, b_proj);
}
